// TopoAwareQIGNN_76794015252986
// MI455X (gfx1250) — matrix operand order repaired and checked by reading, not compiled, not run on hardware
//
#include <hip/hip_runtime.h>
#include <hip/hip_bf16.h>
#include <stdint.h>

#define B_ 64
#define N_ 512
#define D_ 128
#define H_ 4
#define HD_ 32
#define MAX_ITER_ 30

typedef __bf16 bf16_t;
typedef __attribute__((ext_vector_type(16))) __bf16 v16b;
typedef __attribute__((ext_vector_type(8)))  __bf16 v8b;
typedef __attribute__((ext_vector_type(8)))  float  v8f;

static __device__ __forceinline__ v8f zero8() {
  v8f z;
#pragma unroll
  for (int i = 0; i < 8; ++i) z[i] = 0.0f;
  return z;
}

// A-operand (16x32 bf16) from row-major buffer, row stride `ld` elements.
// Lanes 0-15 (g=0) hold K=0..7 & 16..23; lanes 16-31 (g=1) K=8..15 & 24..31.
static __device__ __forceinline__ v16b load_a16x32(const bf16_t* base, int ld) {
  int lane = threadIdx.x & 31;
  int m = lane & 15;
  int g = (lane >> 4) & 1;
  const v8b* p0 = (const v8b*)(base + (size_t)m * ld + g * 8);
  const v8b* p1 = (const v8b*)(base + (size_t)m * ld + 16 + g * 8);
  v8b lo = *p0, hi = *p1;
  return __builtin_shufflevector(lo, hi, 0,1,2,3,4,5,6,7,8,9,10,11,12,13,14,15);
}

// B-operand (32x16 bf16): each output-column's 32 K-values contiguous; column stride `cld`.
static __device__ __forceinline__ v16b load_b32x16(const bf16_t* base, int cld) {
  int lane = threadIdx.x & 31;
  int n = lane & 15;
  int g = (lane >> 4) & 1;
  const v8b* p0 = (const v8b*)(base + (size_t)n * cld + g * 8);
  const v8b* p1 = (const v8b*)(base + (size_t)n * cld + 16 + g * 8);
  v8b lo = *p0, hi = *p1;
  return __builtin_shufflevector(lo, hi, 0,1,2,3,4,5,6,7,8,9,10,11,12,13,14,15);
}

static __device__ __forceinline__ v8f wmma_bf16(v16b a, v16b b, v8f c) {
  return __builtin_amdgcn_wmma_f32_16x16x32_bf16(false, a, false, b, (short)0, c, false, false);
}

// ---------------- prep: power iteration + weight bf16 conversions ----------------
__global__ void prep_kernel(const float* __restrict__ W, const float* __restrict__ w_u,
                            const float* __restrict__ Omega, const float* __restrict__ Wq,
                            const float* __restrict__ Wk, const float* __restrict__ Wv,
                            bf16_t* Weffb, bf16_t* Omegab, bf16_t* Wqb, bf16_t* Wkb, bf16_t* Wvb) {
  __shared__ float su[D_], sv[D_], red[D_];
  int t = threadIdx.x;
  su[t] = w_u[t];
  __syncthreads();
  for (int it = 0; it < 5; ++it) {
    float acc = 0.f;
    for (int k = 0; k < D_; ++k) acc += W[k * D_ + t] * su[k];
    red[t] = acc * acc; __syncthreads();
    for (int s = 64; s > 0; s >>= 1) { if (t < s) red[t] += red[t + s]; __syncthreads(); }
    float nv = sqrtf(red[0]) + 1e-12f; __syncthreads();
    sv[t] = acc / nv; __syncthreads();
    float accu = 0.f;
    for (int k = 0; k < D_; ++k) accu += W[t * D_ + k] * sv[k];
    red[t] = accu * accu; __syncthreads();
    for (int s = 64; s > 0; s >>= 1) { if (t < s) red[t] += red[t + s]; __syncthreads(); }
    float nu = sqrtf(red[0]) + 1e-12f; __syncthreads();
    su[t] = accu / nu; __syncthreads();
  }
  float wv_t = 0.f;
  for (int k = 0; k < D_; ++k) wv_t += W[t * D_ + k] * sv[k];
  red[t] = su[t] * wv_t; __syncthreads();
  for (int s = 64; s > 0; s >>= 1) { if (t < s) red[t] += red[t + s]; __syncthreads(); }
  float sigma = red[0];
  float scale = fminf(1.0f, 0.999f / fmaxf(sigma, 1e-12f));
  for (int i = t; i < D_ * D_; i += D_) {
    Weffb[i]  = (bf16_t)(W[i] * scale);
    Omegab[i] = (bf16_t)Omega[i];
    Wqb[i]    = (bf16_t)Wq[i];
    Wkb[i]    = (bf16_t)Wk[i];
    Wvb[i]    = (bf16_t)Wv[i];
  }
}

// ---------------- f32 -> bf16 conversion ----------------
__global__ void cvt_f32_bf16(const float* __restrict__ src, bf16_t* __restrict__ dst, int n) {
  int i = blockIdx.x * blockDim.x + threadIdx.x;
  int stride = gridDim.x * blockDim.x;
  for (; i < n; i += stride) dst[i] = (bf16_t)src[i];
}

// ---------------- inj = Xb @ Wb^T + bias ----------------
__global__ void inj_kernel(const bf16_t* __restrict__ Xb, const bf16_t* __restrict__ Wb,
                           const float* __restrict__ bias, float* __restrict__ outF) {
  int wv = threadIdx.x >> 5;
  int rt = wv >> 1, ch = wv & 1;
  int lane = threadIdx.x & 31;
  int col16 = lane & 15, rhalf = (lane >> 4) * 8;
  size_t row0 = (size_t)blockIdx.x * 64 + rt * 16;
  v8f acc[4];
#pragma unroll
  for (int c = 0; c < 4; ++c) acc[c] = zero8();
#pragma unroll
  for (int k0 = 0; k0 < D_; k0 += 32) {
    v16b a = load_a16x32(Xb + row0 * D_ + k0, D_);
#pragma unroll
    for (int c = 0; c < 4; ++c) {
      int n0 = (ch * 4 + c) * 16;
      v16b b = load_b32x16(Wb + n0 * D_ + k0, D_);
      acc[c] = wmma_bf16(a, b, acc[c]);
    }
  }
#pragma unroll
  for (int c = 0; c < 4; ++c) {
    int n0 = (ch * 4 + c) * 16;
#pragma unroll
    for (int r = 0; r < 8; ++r) {
      size_t row = row0 + r + rhalf;
      int colg = n0 + col16;
      outF[row * D_ + colg] = acc[c][r] + bias[colg];
    }
  }
}

// async copy of one K-chunk of Zt (128 features x 64 nodes, bf16 = 16KB) into LDS.
// 256 threads, each issues 4 x b128 async loads (ASYNCcnt-tracked).
static __device__ __forceinline__ void async_copy_chunk(const bf16_t* gsrc,   // Zt_b + chunk*64
                                                        unsigned lds_base, int tid) {
#pragma unroll
  for (int s = 0; s < 4; ++s) {
    int seg = tid + s * 256;          // 1024 x 16B segments
    int row = seg >> 3;               // feature 0..127
    int off = (seg & 7) * 16;         // byte offset within 128B row
    const char* g = (const char*)(gsrc + (size_t)row * N_) + off;
    unsigned l = lds_base + (unsigned)(row * 128 + off);
    asm volatile("global_load_async_to_lds_b128 %0, %1, off"
                 :: "v"(l), "v"(g)
                 : "memory");
  }
}

// ---------------- one fixed-point iteration ----------------
__global__ void iter_kernel(const bf16_t* __restrict__ adjb, const bf16_t* __restrict__ Ztin,
                            bf16_t* __restrict__ Ztout, float* __restrict__ Zf,
                            const float* __restrict__ injf, const bf16_t* __restrict__ Weffb,
                            const int* __restrict__ counts) {
  __shared__ __align__(16) bf16_t ldsZ[2][128 * 64];   // double-buffered B slab (2x16KB)
  __shared__ __align__(16) bf16_t ldsT[64 * D_];       // inter-GEMM staging (16KB)
  int rt64 = blockIdx.x;
  int b = rt64 >> 3;
  int r0 = (rt64 & 7) * 64;
  int tid = threadIdx.x;
  int wv = tid >> 5;
  int rt = wv >> 1, ch = wv & 1;
  int lane = tid & 31;
  int col16 = lane & 15, rhalf = (lane >> 4) * 8;

  const bf16_t* Arow = adjb + ((size_t)b * N_ + r0 + rt * 16) * N_;
  const bf16_t* Zt_b = Ztin + (size_t)b * D_ * N_;
  unsigned ldsZ0 = (unsigned)(uintptr_t)(&ldsZ[0][0]);
  unsigned ldsZ1 = (unsigned)(uintptr_t)(&ldsZ[1][0]);

  // GEMM1: T[64x128] = adj[b, r0:r0+64, :] @ Z[b], K = 512 in 8 chunks of 64,
  // B slab staged via async-to-LDS, one chunk ahead (overlaps DMA with WMMA).
  v8f acc[4];
#pragma unroll
  for (int c = 0; c < 4; ++c) acc[c] = zero8();

  async_copy_chunk(Zt_b, ldsZ0, tid);
  for (int chunk = 0; chunk < 8; ++chunk) {
    asm volatile("s_wait_asynccnt 0x0" ::: "memory");   // own chunk DMA complete
    __syncthreads();                                    // all waves' DMA visible; prev readers done
    if (chunk + 1 < 8)
      async_copy_chunk(Zt_b + (chunk + 1) * 64,
                       ((chunk + 1) & 1) ? ldsZ1 : ldsZ0, tid);
    const bf16_t* bufp = &ldsZ[chunk & 1][0];
    if (chunk + 1 < 8) __builtin_prefetch((const void*)(Arow + (chunk + 1) * 64), 0, 1);
#pragma unroll
    for (int kk = 0; kk < 64; kk += 32) {
      v16b a = load_a16x32(Arow + chunk * 64 + kk, N_);
#pragma unroll
      for (int c = 0; c < 4; ++c) {
        int n0 = (ch * 4 + c) * 16;
        v16b bm = load_b32x16(bufp + n0 * 64 + kk, 64);
        acc[c] = wmma_bf16(a, bm, acc[c]);
      }
    }
  }

  // stage T as bf16 in LDS (row-major 64x128)
  __syncthreads();
#pragma unroll
  for (int c = 0; c < 4; ++c) {
    int n0 = (ch * 4 + c) * 16;
#pragma unroll
    for (int r = 0; r < 8; ++r)
      ldsT[(rt * 16 + r + rhalf) * D_ + n0 + col16] = (bf16_t)acc[c][r];
  }
  __syncthreads();

  // GEMM2: pre = T @ Weff^T (K = 128)
  v8f acc2[4];
#pragma unroll
  for (int c = 0; c < 4; ++c) acc2[c] = zero8();
#pragma unroll
  for (int k0 = 0; k0 < D_; k0 += 32) {
    v16b a = load_a16x32(ldsT + (rt * 16) * D_ + k0, D_);
#pragma unroll
    for (int c = 0; c < 4; ++c) {
      int n0 = (ch * 4 + c) * 16;
      v16b bm = load_b32x16(Weffb + n0 * D_ + k0, D_);
      acc2[c] = wmma_bf16(a, bm, acc2[c]);
    }
  }

  int cnt = counts[b] < 1 ? 1 : counts[b];
#pragma unroll
  for (int c = 0; c < 4; ++c) {
    int n0 = (ch * 4 + c) * 16;
#pragma unroll
    for (int r = 0; r < 8; ++r) {
      int rown = r0 + rt * 16 + r + rhalf;
      int colg = n0 + col16;
      size_t gidx = ((size_t)b * N_ + rown) * D_ + colg;
      float pre = acc2[c][r] + injf[gidx];
      float zn = (rown < cnt) ? tanhf(pre) : 0.0f;
      float z = 0.5f * Zf[gidx] + 0.5f * zn;
      Zf[gidx] = z;
      Ztout[((size_t)b * D_ + colg) * N_ + rown] = (bf16_t)z;
    }
  }
}

// ---------------- Q,K (row-major bf16) and V (transposed bf16) projections ----------------
__global__ void qkv_kernel(const bf16_t* __restrict__ Zb, const bf16_t* __restrict__ Wqb,
                           const bf16_t* __restrict__ Wkb, const bf16_t* __restrict__ Wvb,
                           const float* __restrict__ bq, const float* __restrict__ bk,
                           const float* __restrict__ bv,
                           bf16_t* __restrict__ Qb, bf16_t* __restrict__ Kb, bf16_t* __restrict__ Vt) {
  int rt64 = blockIdx.x;
  int b = rt64 >> 3;
  int r0 = (rt64 & 7) * 64;
  int wv = threadIdx.x >> 5;
  int rt = wv >> 1, ch = wv & 1;
  int lane = threadIdx.x & 31;
  int col16 = lane & 15, rhalf = (lane >> 4) * 8;
  const bf16_t* Ws[3] = {Wqb, Wkb, Wvb};
  const float*  bs[3] = {bq, bk, bv};
  const bf16_t* Xrow = Zb + ((size_t)b * N_ + r0 + rt * 16) * D_;
  for (int w = 0; w < 3; ++w) {
    v8f acc[4];
#pragma unroll
    for (int c = 0; c < 4; ++c) acc[c] = zero8();
#pragma unroll
    for (int k0 = 0; k0 < D_; k0 += 32) {
      v16b a = load_a16x32(Xrow + k0, D_);
#pragma unroll
      for (int c = 0; c < 4; ++c) {
        int n0 = (ch * 4 + c) * 16;
        v16b bm = load_b32x16(Ws[w] + n0 * D_ + k0, D_);
        acc[c] = wmma_bf16(a, bm, acc[c]);
      }
    }
#pragma unroll
    for (int c = 0; c < 4; ++c) {
      int n0 = (ch * 4 + c) * 16;
#pragma unroll
      for (int r = 0; r < 8; ++r) {
        int rown = r0 + rt * 16 + r + rhalf;
        int colg = n0 + col16;
        float val = acc[c][r] + bs[w][colg];
        if (w == 0)      Qb[((size_t)b * N_ + rown) * D_ + colg] = (bf16_t)val;
        else if (w == 1) Kb[((size_t)b * N_ + rown) * D_ + colg] = (bf16_t)val;
        else             Vt[((size_t)b * D_ + colg) * N_ + rown] = (bf16_t)val;
      }
    }
  }
}

// ---------------- flash-style masked attention + pooled partial sums ----------------
__global__ void attn_kernel(const bf16_t* __restrict__ Qb, const bf16_t* __restrict__ Kb,
                            const bf16_t* __restrict__ Vt, const int* __restrict__ counts,
                            float* __restrict__ part) {
  __shared__ __align__(16) bf16_t ldsP[8 * 16 * 32];
  __shared__ float spool[8][32];
  int bid = blockIdx.x;                 // B*H*(N/128) = 1024
  int b = bid >> 4;
  int rem = bid & 15;
  int h = rem >> 2;
  int qt = rem & 3;
  int q0 = qt * 128;
  int wv = threadIdx.x >> 5;
  int lane = threadIdx.x & 31;
  int col16 = lane & 15, rhalf = (lane >> 4) * 8;
  int qbase = q0 + wv * 16;
  int cnt = counts[b] < 1 ? 1 : counts[b];
  const float scale = 0.17677669529663687f;   // 1/sqrt(32)

  v16b aQ = load_a16x32(Qb + ((size_t)b * N_ + qbase) * D_ + h * HD_, D_);
  v8f o0 = zero8(), o1 = zero8();
  float mrow[8], lrow[8];
#pragma unroll
  for (int r = 0; r < 8; ++r) { mrow[r] = -1e30f; lrow[r] = 0.0f; }

  bf16_t* myP = ldsP + wv * (16 * 32);
  int nkt = (cnt + 31) >> 5;
  for (int kt = 0; kt < nkt; ++kt) {
    int k0 = kt * 32;
    v16b bK0 = load_b32x16(Kb + ((size_t)b * N_ + k0) * D_ + h * HD_, D_);
    v16b bK1 = load_b32x16(Kb + ((size_t)b * N_ + k0 + 16) * D_ + h * HD_, D_);
    v8f s0 = wmma_bf16(aQ, bK0, zero8());
    v8f s1 = wmma_bf16(aQ, bK1, zero8());
    float p0[8], p1[8], mx[8];
    bool v0 = (k0 + col16) < cnt;
    bool v1 = (k0 + 16 + col16) < cnt;
#pragma unroll
    for (int r = 0; r < 8; ++r) {
      p0[r] = v0 ? s0[r] * scale : -1e9f;
      p1[r] = v1 ? s1[r] * scale : -1e9f;
      mx[r] = fmaxf(p0[r], p1[r]);
    }
#pragma unroll
    for (int off = 1; off < 16; off <<= 1)
#pragma unroll
      for (int r = 0; r < 8; ++r) mx[r] = fmaxf(mx[r], __shfl_xor(mx[r], off, 32));
    float corr[8], rs[8];
#pragma unroll
    for (int r = 0; r < 8; ++r) {
      float mnew = fmaxf(mrow[r], mx[r]);
      corr[r] = __expf(mrow[r] - mnew);
      p0[r] = __expf(p0[r] - mnew);
      p1[r] = __expf(p1[r] - mnew);
      mrow[r] = mnew;
      rs[r] = p0[r] + p1[r];
    }
#pragma unroll
    for (int off = 1; off < 16; off <<= 1)
#pragma unroll
      for (int r = 0; r < 8; ++r) rs[r] += __shfl_xor(rs[r], off, 32);
#pragma unroll
    for (int r = 0; r < 8; ++r) {
      lrow[r] = lrow[r] * corr[r] + rs[r];
      o0[r] *= corr[r];
      o1[r] *= corr[r];
      int row = r + rhalf;
      myP[row * 32 + col16]      = (bf16_t)p0[r];
      myP[row * 32 + 16 + col16] = (bf16_t)p1[r];
    }
    v16b aP  = load_a16x32(myP, 32);
    v16b bV0 = load_b32x16(Vt + ((size_t)b * D_ + h * HD_ + 0)  * N_ + k0, N_);
    v16b bV1 = load_b32x16(Vt + ((size_t)b * D_ + h * HD_ + 16) * N_ + k0, N_);
    o0 = wmma_bf16(aP, bV0, o0);
    o1 = wmma_bf16(aP, bV1, o1);
  }

  float cs0 = 0.f, cs1 = 0.f;
#pragma unroll
  for (int r = 0; r < 8; ++r) {
    int qrow = qbase + r + rhalf;
    float qm = (qrow < cnt) ? 1.0f : 0.0f;
    float inv = qm / lrow[r];
    cs0 += o0[r] * inv;
    cs1 += o1[r] * inv;
  }
  cs0 += __shfl_xor(cs0, 16, 32);
  cs1 += __shfl_xor(cs1, 16, 32);
  if (lane < 16) {
    spool[wv][lane]      = cs0;
    spool[wv][16 + lane] = cs1;
  }
  __syncthreads();
  if (threadIdx.x < 32) {
    float s = 0.f;
#pragma unroll
    for (int w = 0; w < 8; ++w) s += spool[w][threadIdx.x];
    part[((size_t)b * 4 + qt) * D_ + h * HD_ + threadIdx.x] = s;   // deterministic partials
  }
}

// ---------------- out = pooled @ Wo^T + bo ----------------
__global__ void out_kernel(const float* __restrict__ part, const float* __restrict__ Wo,
                           const float* __restrict__ bo, float* __restrict__ out) {
  __shared__ float sp[D_];
  int b = blockIdx.x;
  int n = threadIdx.x;
  sp[n] = part[((size_t)b * 4 + 0) * D_ + n] + part[((size_t)b * 4 + 1) * D_ + n] +
          part[((size_t)b * 4 + 2) * D_ + n] + part[((size_t)b * 4 + 3) * D_ + n];
  __syncthreads();
  float acc = bo[n];
  for (int k = 0; k < D_; ++k) acc += sp[k] * Wo[n * D_ + k];
  out[b * D_ + n] = acc;
}

extern "C" void kernel_launch(void* const* d_in, const int* in_sizes, int n_in,
                              void* d_out, int out_size, void* d_ws, size_t ws_size,
                              hipStream_t stream) {
  const float* h      = (const float*)d_in[0];
  const float* adj    = (const float*)d_in[1];
  const int*   counts = (const int*)d_in[2];
  const float* W      = (const float*)d_in[3];
  const float* Omega  = (const float*)d_in[4];
  const float* bias   = (const float*)d_in[5];
  const float* w_u    = (const float*)d_in[6];
  const float* Wq = (const float*)d_in[7];  const float* bq = (const float*)d_in[8];
  const float* Wk = (const float*)d_in[9];  const float* bk = (const float*)d_in[10];
  const float* Wv = (const float*)d_in[11]; const float* bv = (const float*)d_in[12];
  const float* Wo = (const float*)d_in[13]; const float* bo = (const float*)d_in[14];
  float* out = (float*)d_out;

  char* ws = (char*)d_ws;
  size_t off = 0;
  auto alloc = [&](size_t bytes) -> void* {
    void* p = ws + off;
    off += (bytes + 255) & ~(size_t)255;
    return p;
  };
  bf16_t* adjb   = (bf16_t*)alloc((size_t)B_ * N_ * N_ * 2);
  float*  injf   = (float*) alloc((size_t)B_ * N_ * D_ * 4);
  float*  Zf     = (float*) alloc((size_t)B_ * N_ * D_ * 4);
  bf16_t* Zt0    = (bf16_t*)alloc((size_t)B_ * N_ * D_ * 2);
  bf16_t* Zt1    = (bf16_t*)alloc((size_t)B_ * N_ * D_ * 2);
  bf16_t* Xb     = (bf16_t*)alloc((size_t)B_ * N_ * D_ * 2);   // h bf16, later Z bf16
  bf16_t* Qb     = (bf16_t*)alloc((size_t)B_ * N_ * D_ * 2);
  bf16_t* Kb     = (bf16_t*)alloc((size_t)B_ * N_ * D_ * 2);
  bf16_t* Vt     = (bf16_t*)alloc((size_t)B_ * N_ * D_ * 2);
  bf16_t* Weffb  = (bf16_t*)alloc((size_t)D_ * D_ * 2);
  bf16_t* Omegab = (bf16_t*)alloc((size_t)D_ * D_ * 2);
  bf16_t* Wqb    = (bf16_t*)alloc((size_t)D_ * D_ * 2);
  bf16_t* Wkb    = (bf16_t*)alloc((size_t)D_ * D_ * 2);
  bf16_t* Wvb    = (bf16_t*)alloc((size_t)D_ * D_ * 2);
  float*  part   = (float*) alloc((size_t)B_ * 4 * D_ * 4);

  hipMemsetAsync(Zf,  0, (size_t)B_ * N_ * D_ * 4, stream);
  hipMemsetAsync(Zt0, 0, (size_t)B_ * N_ * D_ * 2, stream);

  prep_kernel<<<1, D_, 0, stream>>>(W, w_u, Omega, Wq, Wk, Wv, Weffb, Omegab, Wqb, Wkb, Wvb);
  cvt_f32_bf16<<<2048, 256, 0, stream>>>(adj, adjb, B_ * N_ * N_);
  cvt_f32_bf16<<<1024, 256, 0, stream>>>(h, Xb, B_ * N_ * D_);

  inj_kernel<<<B_ * N_ / 64, 256, 0, stream>>>(Xb, Omegab, bias, injf);

  for (int it = 0; it < MAX_ITER_; ++it) {
    bf16_t* zin  = (it & 1) ? Zt1 : Zt0;
    bf16_t* zout = (it & 1) ? Zt0 : Zt1;
    iter_kernel<<<B_ * N_ / 64, 256, 0, stream>>>(adjb, zin, zout, Zf, injf, Weffb, counts);
  }

  cvt_f32_bf16<<<1024, 256, 0, stream>>>(Zf, Xb, B_ * N_ * D_);
  qkv_kernel<<<B_ * N_ / 64, 256, 0, stream>>>(Xb, Wqb, Wkb, Wvb, bq, bk, bv, Qb, Kb, Vt);
  attn_kernel<<<B_ * H_ * (N_ / 128), 256, 0, stream>>>(Qb, Kb, Vt, counts, part);
  out_kernel<<<B_, D_, 0, stream>>>(part, Wo, bo, out);
}
